// MultiHeadAttention_14491219656894
// MI455X (gfx1250) — matrix operand order repaired and checked by reading, not compiled, not run on hardware
//
#include <hip/hip_runtime.h>

// MI455X / gfx1250, wave32. Fused MHA w/ relative position bias.
// f16 operands everywhere (pre-converted once), f32 WMMA accumulation.
// Round 3: branch-free templated GEMM epilogue (no per-element 64-bit muls).

typedef __attribute__((ext_vector_type(16))) _Float16 v16h;
typedef __attribute__((ext_vector_type(8)))  _Float16 h8;
typedef __attribute__((ext_vector_type(8)))  float    v8f;

#define HEADS   16
#define DH      64
#define LSEQ    1024
#define DMODEL  1024
#define MAXLEN  150
#define NPOS    301           // 2*MAXLEN+1
#define NEG_INF (-1e30f)

static __device__ __forceinline__ v8f zero8() {
  v8f z;
#pragma unroll
  for (int i = 0; i < 8; ++i) z[i] = 0.0f;
  return z;
}
static __device__ __forceinline__ v16h zero16h() {
  v16h z;
#pragma unroll
  for (int i = 0; i < 16; ++i) z[i] = (_Float16)0.0f;
  return z;
}
static __device__ __forceinline__ v16h join16(h8 lo, h8 hi) {
  return __builtin_shufflevector(lo, hi, 0, 1, 2, 3, 4, 5, 6, 7,
                                 8, 9, 10, 11, 12, 13, 14, 15);
}

// A-fragment (16x32 f16, M x K), f16 source. Lane l: row m = l&15, kg = l>>4.
// halves i<8 -> K = 8*kg + i ; i>=8 -> K = 16 + 8*kg + (i-8)
static __device__ __forceinline__ v16h load_a_frag(const _Float16* __restrict__ rowp, int kg) {
  h8 lo = *(const h8*)(rowp + 8 * kg);
  h8 hi = *(const h8*)(rowp + 16 + 8 * kg);
  return join16(lo, hi);
}

// B-fragment (32x16 f16, K x N) where B[K,n] = row_n[K]. Lane l: n = l&15.
// halves i<8 -> K = 8*kg + i ; i>=8 -> K = 16 + 8*kg + (i-8)
static __device__ __forceinline__ v16h load_b_frag(const _Float16* __restrict__ rowp, int kg) {
  h8 lo = *(const h8*)(rowp + 8 * kg);
  h8 hi = *(const h8*)(rowp + 16 + 8 * kg);
  return join16(lo, hi);
}

// ---------------------------------------------------------------------------
// Elementwise f32 -> f16 (8 elements / thread, b128 in / b128 out).
// ---------------------------------------------------------------------------
__global__ __launch_bounds__(256)
void f32_to_f16_kernel(const float* __restrict__ src, _Float16* __restrict__ dst, int n8) {
  int i = blockIdx.x * 256 + threadIdx.x;
  if (i >= n8) return;
  float4 a = ((const float4*)src)[2 * i];
  float4 b = ((const float4*)src)[2 * i + 1];
  h8 o;
  o[0] = (_Float16)a.x; o[1] = (_Float16)a.y; o[2] = (_Float16)a.z; o[3] = (_Float16)a.w;
  o[4] = (_Float16)b.x; o[5] = (_Float16)b.y; o[6] = (_Float16)b.z; o[7] = (_Float16)b.w;
  ((h8*)dst)[i] = o;
}

// ---------------------------------------------------------------------------
// Y[M,N] = Xh[M,K] @ Wh[N,K]^T + bias[N]    (nn.Linear semantics, f16 in)
// One wave per block owns a 32(M) x 64(N) tile: 8 WMMA vs 12 b128 loads per
// K-step.  OUT_F32 selects the output element type at compile time.
// ---------------------------------------------------------------------------
template <bool OUT_F32>
__global__ __launch_bounds__(32)
void gemm_bias_f16(const _Float16* __restrict__ Xh, const _Float16* __restrict__ Wh,
                   const float* __restrict__ bias, void* __restrict__ Yout,
                   int M, int N, int K) {
  const int lane = threadIdx.x & 31;
  const int ln = lane & 15;
  const int kg = lane >> 4;
  const int mBase = blockIdx.x * 32;
  const int nBase = blockIdx.y * 64;

  v8f c[2][4];
#pragma unroll
  for (int u = 0; u < 2; ++u)
#pragma unroll
    for (int t = 0; t < 4; ++t) c[u][t] = zero8();

  const _Float16* xrow0 = Xh + (size_t)(mBase + ln) * K;
  const _Float16* xrow1 = Xh + (size_t)(mBase + 16 + ln) * K;

  for (int kb = 0; kb < K; kb += 32) {
    __builtin_prefetch(xrow0 + kb + 256, 0, 1);        // global_prefetch_b8
    // issue all 12 fragment loads up front, then 8 WMMAs
    v16h a0 = load_a_frag(xrow0 + kb, kg);
    v16h a1 = load_a_frag(xrow1 + kb, kg);
    v16h bf[4];
#pragma unroll
    for (int t = 0; t < 4; ++t)
      bf[t] = load_b_frag(Wh + (size_t)(nBase + 16 * t + ln) * K + kb, kg);
#pragma unroll
    for (int t = 0; t < 4; ++t)
      c[0][t] = __builtin_amdgcn_wmma_f32_16x16x32_f16(false, a0, false, bf[t],
                                                       (short)0, c[0][t], false, false);
#pragma unroll
    for (int t = 0; t < 4; ++t)
      c[1][t] = __builtin_amdgcn_wmma_f32_16x16x32_f16(false, a1, false, bf[t],
                                                       (short)0, c[1][t], false, false);
  }

  // branch-free epilogue: base pointer per (u,t) column, step N per row
#pragma unroll
  for (int u = 0; u < 2; ++u) {
    const size_t rowBase = (size_t)(mBase + 16 * u + 8 * kg) * N;
#pragma unroll
    for (int t = 0; t < 4; ++t) {
      const int n = nBase + 16 * t + ln;
      const float bv = bias[n];
      if (OUT_F32) {
        float* p = (float*)Yout + rowBase + n;
#pragma unroll
        for (int r = 0; r < 8; ++r) { *p = c[u][t][r] + bv; p += N; }
      } else {
        _Float16* p = (_Float16*)Yout + rowBase + n;
#pragma unroll
        for (int r = 0; r < 8; ++r) { *p = (_Float16)(c[u][t][r] + bv); p += N; }
      }
    }
  }
}

// ---------------------------------------------------------------------------
// Flash attention with relative-position bias (all-f16 operands, f32 accum).
// grid = (L/16, H, B); 1 wave per block handles 16 query rows for one (b,h).
// P[q,r] = q . pos_emb[r] (19 WMMA tiles) cached in LDS; scores gather
// P[q, idx(q,k)] instead of an L x L einsum.
// ---------------------------------------------------------------------------
__global__ __launch_bounds__(32)
void attn_rel_wmma(const _Float16* __restrict__ q_h, const _Float16* __restrict__ k_h,
                   const _Float16* __restrict__ v_h, const int* __restrict__ mask,
                   const _Float16* __restrict__ pos_h, _Float16* __restrict__ attn_h) {
  const int lane = threadIdx.x & 31;
  const int ln = lane & 15;
  const int kg = lane >> 4;
  const int q0 = blockIdx.x * 16;
  const int h  = blockIdx.y;
  const int b  = blockIdx.z;

  __shared__ float    P[16][NPOS + 3];   // rel-pos scores (padded to 304 cols)
  __shared__ _Float16 vT[DH][34];        // V chunk transposed: vT[d][k], padded
  __shared__ _Float16 pT[16][40];        // prob tile 16x32 f16, padded

  // --- q tile A-fragments (d-chunks 0..31 and 32..63), loaded once ---
  const _Float16* qrow = q_h + ((size_t)(b * LSEQ + q0 + ln)) * DMODEL + h * DH;
  const v16h qa0 = load_a_frag(qrow, kg);
  const v16h qa1 = load_a_frag(qrow + 32, kg);

  // --- P = q_tile @ pos_emb^T  (cols padded to 304) ---
  for (int rt = 0; rt < 19; ++rt) {
    const int r = rt * 16 + ln;
    v16h b0 = zero16h(), b1 = zero16h();
    if (r < NPOS) {
      const _Float16* prow = pos_h + (size_t)r * DH;
      b0 = load_b_frag(prow, kg);
      b1 = load_b_frag(prow + 32, kg);
    }
    v8f acc = zero8();
    acc = __builtin_amdgcn_wmma_f32_16x16x32_f16(false, qa0, false, b0, (short)0, acc, false, false);
    acc = __builtin_amdgcn_wmma_f32_16x16x32_f16(false, qa1, false, b1, (short)0, acc, false, false);
#pragma unroll
    for (int r8 = 0; r8 < 8; ++r8) P[r8 + 8 * kg][rt * 16 + ln] = acc[r8];
  }

  // --- online softmax state (each half-wave holds rows m = r + 8*kg) ---
  float m_run[8], l_run[8];
  v8f o[4];
#pragma unroll
  for (int r = 0; r < 8; ++r) { m_run[r] = -3.0e38f; l_run[r] = 0.0f; }
#pragma unroll
  for (int t = 0; t < 4; ++t) o[t] = zero8();

  for (int kb = 0; kb < LSEQ; kb += 32) {
    // ---- scores for two 16-key tiles: load all 4 K-fragments, then 4 WMMAs ----
    const _Float16* krow0 = k_h + ((size_t)(b * LSEQ + kb + ln)) * DMODEL + h * DH;
    const _Float16* krow1 = krow0 + (size_t)16 * DMODEL;
    v16h kf[2][2];
    kf[0][0] = load_b_frag(krow0, kg);
    kf[0][1] = load_b_frag(krow0 + 32, kg);
    kf[1][0] = load_b_frag(krow1, kg);
    kf[1][1] = load_b_frag(krow1 + 32, kg);

    v8f s[2];
#pragma unroll
    for (int t2 = 0; t2 < 2; ++t2) {
      v8f acc = zero8();
      acc = __builtin_amdgcn_wmma_f32_16x16x32_f16(false, qa0, false, kf[t2][0], (short)0, acc, false, false);
      acc = __builtin_amdgcn_wmma_f32_16x16x32_f16(false, qa1, false, kf[t2][1], (short)0, acc, false, false);
      const int kcol = kb + t2 * 16 + ln;
      const int* mrow = mask + (size_t)(b * LSEQ + q0 + 8 * kg) * LSEQ + kcol;
#pragma unroll
      for (int r = 0; r < 8; ++r) {
        const int m = r + 8 * kg;
        const int qAbs = q0 + m;
        int rel = kcol - qAbs;
        rel = rel < -MAXLEN ? -MAXLEN : (rel > MAXLEN ? MAXLEN : rel);
        int idx = rel + MAXLEN - 1;          // python: (rel + 149) % 301
        if (idx < 0) idx += NPOS;
        float sc = (acc[r] + P[m][idx]) * 0.125f;   // 1/sqrt(64)
        if (mrow[(size_t)r * LSEQ]) sc = NEG_INF;
        acc[r] = sc;
      }
      s[t2] = acc;
    }

    // ---- online softmax update (reduce across 16 lanes of each half) ----
    float alpha[8];
#pragma unroll
    for (int r = 0; r < 8; ++r) {
      float mx = fmaxf(s[0][r], s[1][r]);
      mx = fmaxf(mx, __shfl_xor(mx, 1, 32));
      mx = fmaxf(mx, __shfl_xor(mx, 2, 32));
      mx = fmaxf(mx, __shfl_xor(mx, 4, 32));
      mx = fmaxf(mx, __shfl_xor(mx, 8, 32));
      const float mnew = fmaxf(m_run[r], mx);
      const float a = __expf(m_run[r] - mnew);
      const float p0 = __expf(s[0][r] - mnew);
      const float p1 = __expf(s[1][r] - mnew);
      s[0][r] = p0; s[1][r] = p1;
      float ps = p0 + p1;
      ps += __shfl_xor(ps, 1, 32);
      ps += __shfl_xor(ps, 2, 32);
      ps += __shfl_xor(ps, 4, 32);
      ps += __shfl_xor(ps, 8, 32);
      l_run[r] = l_run[r] * a + ps;
      m_run[r] = mnew;
      alpha[r] = a;
    }
#pragma unroll
    for (int t = 0; t < 4; ++t)
#pragma unroll
      for (int r = 0; r < 8; ++r) o[t][r] *= alpha[r];

    // ---- restage probs (C-layout -> A-layout via LDS) ----
#pragma unroll
    for (int t2 = 0; t2 < 2; ++t2)
#pragma unroll
      for (int r = 0; r < 8; ++r)
        pT[r + 8 * kg][t2 * 16 + ln] = (_Float16)s[t2][r];

    // ---- stage V chunk transposed into LDS (one key row per lane, f16 in) ----
    const _Float16* vrow = v_h + ((size_t)(b * LSEQ + kb + lane)) * DMODEL + h * DH;
#pragma unroll
    for (int j = 0; j < 8; ++j) {
      h8 vv = *(const h8*)(vrow + 8 * j);
#pragma unroll
      for (int e = 0; e < 8; ++e) vT[8 * j + e][lane] = vv[e];
    }

    // ---- A-fragment from prob tile ----
    v16h pa;
#pragma unroll
    for (int i = 0; i < 16; ++i) {
      const int kk = i + (i >= 8 ? 8 : 0) + 8 * kg;    // A layout K index
      pa[i] = pT[ln][kk];
    }

    // ---- PV: o[16 x 64] += p[16 x 32] @ V[32 x 64] ----
#pragma unroll
    for (int t = 0; t < 4; ++t) {
      v16h vb;
#pragma unroll
      for (int i = 0; i < 16; ++i) vb[i] = vT[16 * t + ln][i + (i >= 8 ? 8 : 0) + 8 * kg];
      o[t] = __builtin_amdgcn_wmma_f32_16x16x32_f16(false, pa, false, vb,
                                                    (short)0, o[t], false, false);
    }
  }

  // ---- normalize and write attn as f16 (feeds final projection GEMM) ----
  _Float16* orow = attn_h + ((size_t)(b * LSEQ + q0 + 8 * kg)) * DMODEL + h * DH + ln;
#pragma unroll
  for (int t = 0; t < 4; ++t) {
    _Float16* p = orow + 16 * t;
#pragma unroll
    for (int r = 0; r < 8; ++r) {
      p[(size_t)r * DMODEL] = (_Float16)(o[t][r] / l_run[r]);
    }
  }
}

// ---------------------------------------------------------------------------
extern "C" void kernel_launch(void* const* d_in, const int* in_sizes, int n_in,
                              void* d_out, int out_size, void* d_ws, size_t ws_size,
                              hipStream_t stream) {
  (void)in_sizes; (void)n_in; (void)out_size; (void)ws_size;
  const float* Q   = (const float*)d_in[0];
  const float* K   = (const float*)d_in[1];
  const float* V   = (const float*)d_in[2];
  const int*   msk = (const int*)d_in[3];     // bool mask delivered as int32
  const float* Wq  = (const float*)d_in[4];
  const float* bq  = (const float*)d_in[5];
  const float* Wk  = (const float*)d_in[6];
  const float* bk  = (const float*)d_in[7];
  const float* Wv  = (const float*)d_in[8];
  const float* bv  = (const float*)d_in[9];
  const float* Wo  = (const float*)d_in[10];
  const float* bo  = (const float*)d_in[11];
  const float* pos = (const float*)d_in[12];
  float* out = (float*)d_out;

  const int Mrows = 4 * LSEQ;                          // 4096
  const size_t nAct = (size_t)Mrows * DMODEL;          // 4M elements
  const size_t nWgt = (size_t)DMODEL * DMODEL;         // 1M elements
  const size_t nPos = (size_t)NPOS * DH;               // 19264 elements

  // f16 workspace layout
  char* ws = (char*)d_ws;
  size_t cur = 0;
  auto alloc_h = [&](size_t nelem) -> _Float16* {
    _Float16* p = (_Float16*)(ws + cur);
    cur += ((nelem * sizeof(_Float16) + 255) & ~(size_t)255);
    return p;
  };
  _Float16* Qh   = alloc_h(nAct);
  _Float16* Kh   = alloc_h(nAct);
  _Float16* Vh   = alloc_h(nAct);
  _Float16* Wqh  = alloc_h(nWgt);
  _Float16* Wkh  = alloc_h(nWgt);
  _Float16* Wvh  = alloc_h(nWgt);
  _Float16* Woh  = alloc_h(nWgt);
  _Float16* posh = alloc_h(nPos);
  _Float16* q_h  = alloc_h(nAct);
  _Float16* k_h  = alloc_h(nAct);
  _Float16* v_h  = alloc_h(nAct);
  _Float16* at_h = alloc_h(nAct);

  auto cvt = [&](const float* s, _Float16* d, size_t n) {
    int n8 = (int)(n / 8);
    f32_to_f16_kernel<<<(n8 + 255) / 256, 256, 0, stream>>>(s, d, n8);
  };
  cvt(Q, Qh, nAct);  cvt(K, Kh, nAct);  cvt(V, Vh, nAct);
  cvt(Wq, Wqh, nWgt); cvt(Wk, Wkh, nWgt); cvt(Wv, Wvh, nWgt); cvt(Wo, Woh, nWgt);
  cvt(pos, posh, nPos);

  dim3 gl(Mrows / 32, DMODEL / 64);
  gemm_bias_f16<false><<<gl, 32, 0, stream>>>(Qh, Wqh, bq, (void*)q_h, Mrows, DMODEL, DMODEL);
  gemm_bias_f16<false><<<gl, 32, 0, stream>>>(Kh, Wkh, bk, (void*)k_h, Mrows, DMODEL, DMODEL);
  gemm_bias_f16<false><<<gl, 32, 0, stream>>>(Vh, Wvh, bv, (void*)v_h, Mrows, DMODEL, DMODEL);

  dim3 ga(LSEQ / 16, HEADS, 4);
  attn_rel_wmma<<<ga, 32, 0, stream>>>(q_h, k_h, v_h, msk, posh, at_h);

  gemm_bias_f16<true><<<gl, 32, 0, stream>>>(at_h, Woh, bo, (void*)out, Mrows, DMODEL, DMODEL);
}
